// LiteLA_GLUMB_Block_58987080843394
// MI455X (gfx1250) — matrix operand order repaired and checked by reading, not compiled, not run on hardware
//
#include <hip/hip_runtime.h>
#include <hip/hip_bf16.h>

// ---------------------------------------------------------------------------
// LiteLA + GLUMBConv block for MI455X (gfx1250, wave32, WMMA bf16).
// GEMMs: v_wmma_f32_16x16x32_bf16 with double-buffered async global->LDS
// staging (GLOBAL_LOAD_ASYNC_TO_LDS_B128, tracked by ASYNCcnt).
// ---------------------------------------------------------------------------

typedef __bf16 bf16;
typedef __attribute__((ext_vector_type(16))) __bf16 v16bf;
typedef __attribute__((ext_vector_type(8)))  __bf16 bfx8;   // 16 bytes
typedef __attribute__((ext_vector_type(4)))  __bf16 bfx4;   // 8 bytes
typedef __attribute__((ext_vector_type(8)))  float  v8f;

#define BBATCH 2
#define NTOK   4096
#define CCH    1024
#define HHH    64
#define WWW    64
#define HEADS  32
#define HDIM   32
#define HID    4096
#define MROWS  (BBATCH * NTOK)   // 8192

#ifndef GEMM_ASYNC
#define GEMM_ASYNC 1             // 1: async global->LDS pipeline, 0: sync staging
#endif

static __device__ __forceinline__ float sigmoidf_(float t) {
    return 1.f / (1.f + __expf(-t));
}

#if GEMM_ASYNC
// Async copy of 16 bytes global -> LDS, tracked by ASYNCcnt (no VGPR data).
static __device__ __forceinline__ void async_b128(unsigned lds_off, const void* gptr) {
    asm volatile("global_load_async_to_lds_b128 %0, %1, off"
                 :: "v"(lds_off), "v"(gptr)
                 : "memory");
}
static __device__ __forceinline__ void wait_async0() {
    asm volatile("s_wait_asynccnt 0" ::: "memory");
}
#endif

// ---------------------------------------------------------------------------
// fp32 -> bf16 elementwise conversion (weights)
// ---------------------------------------------------------------------------
__global__ void f32_to_bf16_kernel(const float* __restrict__ in,
                                   bf16* __restrict__ out, long n) {
    long i = (long)blockIdx.x * blockDim.x + threadIdx.x;
    long stride = (long)gridDim.x * blockDim.x;
    for (; i < n; i += stride) out[i] = (bf16)in[i];
}

// ---------------------------------------------------------------------------
// RMSNorm over C=1024, scale by weight, emit bf16. One block (256 thr) per row.
// ---------------------------------------------------------------------------
__global__ __launch_bounds__(256)
void rmsnorm_bf16_kernel(const float* __restrict__ x, const float* __restrict__ w,
                         bf16* __restrict__ out) {
    __shared__ float red[256];
    int row = blockIdx.x, t = threadIdx.x;
    const float* xr = x + (size_t)row * CCH;
    float v[4], s = 0.f;
    for (int j = 0; j < 4; ++j) { v[j] = xr[t + j * 256]; s += v[j] * v[j]; }
    red[t] = s; __syncthreads();
    for (int o = 128; o > 0; o >>= 1) {
        if (t < o) red[t] += red[t + o];
        __syncthreads();
    }
    float scale = rsqrtf(red[0] * (1.f / CCH) + 1e-6f);
    bf16* orow = out + (size_t)row * CCH;
    for (int j = 0; j < 4; ++j) {
        int c = t + j * 256;
        orow[c] = (bf16)(v[j] * scale * w[c]);
    }
}

// ---------------------------------------------------------------------------
// q/k RMSNorm + ReLU + v pass-through, all emitted as bf16.
// ---------------------------------------------------------------------------
__global__ __launch_bounds__(256)
void qknorm_kernel(const float* __restrict__ qkv, const float* __restrict__ qnw,
                   const float* __restrict__ knw,
                   bf16* __restrict__ qr, bf16* __restrict__ kr,
                   bf16* __restrict__ vb) {
    __shared__ float redq[256], redk[256];
    int row = blockIdx.x, t = threadIdx.x;
    const float* base = qkv + (size_t)row * (3 * CCH);
    float qv[4], kv4[4], vv[4], sq = 0.f, sk = 0.f;
    for (int j = 0; j < 4; ++j) {
        int c = t + j * 256;
        qv[j]  = base[c];
        kv4[j] = base[CCH + c];
        vv[j]  = base[2 * CCH + c];
        sq += qv[j] * qv[j];
        sk += kv4[j] * kv4[j];
    }
    redq[t] = sq; redk[t] = sk; __syncthreads();
    for (int o = 128; o > 0; o >>= 1) {
        if (t < o) { redq[t] += redq[t + o]; redk[t] += redk[t + o]; }
        __syncthreads();
    }
    float scq = rsqrtf(redq[0] * (1.f / CCH) + 1e-6f);
    float sck = rsqrtf(redk[0] * (1.f / CCH) + 1e-6f);
    size_t ro = (size_t)row * CCH;
    for (int j = 0; j < 4; ++j) {
        int c = t + j * 256;
        qr[ro + c] = (bf16)fmaxf(qv[j] * scq * qnw[c], 0.f);
        kr[ro + c] = (bf16)fmaxf(kv4[j] * sck * knw[c], 0.f);
        vb[ro + c] = (bf16)vv[j];
    }
}

// ---------------------------------------------------------------------------
// Tiled WMMA bf16 GEMM:  C[M,N] = A[M,K] @ Bw[N,K]^T   (+ epilogue)
//   EPI 0 : fp32 out
//   EPI 1 : fp32 out + optional bias + residual
//   EPI 2 : silu(acc + bias) -> bf16 out
// Block = 256 thr (8 wave32), tile 128x128x64, double-buffered LDS,
// async global->LDS staging. 16 WMMAs per stage, one barrier per stage.
// Fragment layouts per CDNA5 ISA 7.12.2 (16-bit A 16x32 / B 32x16).
// ---------------------------------------------------------------------------
#define TBM 128
#define TBN 128
#define TBK 64
#define LDA 72                      // padded row length (elems) -> 144B stride
#define TILEB (TBM * LDA * 2)       // bytes per (matrix, buffer) = 18432

template<int EPI>
__global__ __launch_bounds__(256)
void gemm_bf16_wmma(const bf16* __restrict__ A, const bf16* __restrict__ Bw,
                    const float* __restrict__ bias, const float* __restrict__ resid,
                    float* __restrict__ Cf, bf16* __restrict__ Cb,
                    int M, int Nn, int K) {
    __shared__ bf16 As[2 * TBM * LDA];
    __shared__ bf16 Bs[2 * TBN * LDA];
    int t = threadIdx.x;
    int wave = t >> 5, lane = t & 31;
    int half = lane >> 4, l16 = lane & 15;
    int rw = wave >> 2, cw = wave & 3;            // 2x4 wave grid -> 64x32 / wave
    int m0 = blockIdx.y * TBM, n0 = blockIdx.x * TBN;

    v8f acc[4][2];
    for (int i = 0; i < 4; ++i)
        for (int j = 0; j < 2; ++j)
            for (int e = 0; e < 8; ++e) acc[i][j][e] = 0.f;

#if GEMM_ASYNC
    unsigned as_base = (unsigned)(uintptr_t)&As[0];   // AS3 offset (low 32 bits)
    unsigned bs_base = (unsigned)(uintptr_t)&Bs[0];
    // Stage a 128x64 bf16 tile of A and B into buffer `buf` (async).
    auto issue_stage = [&](int buf, int k0) {
        for (int i = 0; i < 4; ++i) {
            int e = (t + i * 256) * 8;            // 0..8184 step 8
            int r = e >> 6, c = e & 63;
            unsigned loff = (unsigned)(r * LDA + c) * 2u + (unsigned)buf * TILEB;
            async_b128(as_base + loff, &A [(size_t)(m0 + r) * K + k0 + c]);
            async_b128(bs_base + loff, &Bw[(size_t)(n0 + r) * K + k0 + c]);
        }
    };
#else
    auto issue_stage = [&](int buf, int k0) {
        for (int i = 0; i < 4; ++i) {
            int e = (t + i * 256) * 8;
            int r = e >> 6, c = e & 63;
            int loff = buf * TBM * LDA + r * LDA + c;
            *(bfx8*)&As[loff] = *(const bfx8*)&A [(size_t)(m0 + r) * K + k0 + c];
            *(bfx8*)&Bs[loff] = *(const bfx8*)&Bw[(size_t)(n0 + r) * K + k0 + c];
        }
    };
#endif

    int ntiles = K / TBK;
    issue_stage(0, 0);
    for (int kt = 0; kt < ntiles; ++kt) {
        int cur = kt & 1;
#if GEMM_ASYNC
        wait_async0();          // buffer `cur` (issued last iteration) is complete
#endif
        __syncthreads();        // ...and all waves finished reading buffer 1-cur
        if (kt + 1 < ntiles) issue_stage(1 - cur, (kt + 1) * TBK);

        const bf16* Ab = &As[cur * TBM * LDA];
        const bf16* Bb = &Bs[cur * TBN * LDA];
        for (int ks = 0; ks < 2; ++ks) {          // two K=32 WMMA steps per stage
            v16bf af[4], bfr[2];
            for (int mt = 0; mt < 4; ++mt) {
                // A 16x32: lanes 0-15 K=0..7/16..23, lanes 16-31 K=8..15/24..31
                const bf16* ap = Ab + (rw * 64 + mt * 16 + l16) * LDA + ks * 32 + half * 8;
                ((bfx8*)&af[mt])[0] = *(const bfx8*)ap;
                ((bfx8*)&af[mt])[1] = *(const bfx8*)(ap + 16);
            }
            for (int nt = 0; nt < 2; ++nt) {
                // B 32x16: lanes 0-15 K=0..7/16..23, lanes 16-31 K=8..15/24..31 (col = l16)
                const bf16* bp = Bb + (cw * 32 + nt * 16 + l16) * LDA + ks * 32 + half * 8;
                ((bfx8*)&bfr[nt])[0] = *(const bfx8*)bp;
                ((bfx8*)&bfr[nt])[1] = *(const bfx8*)(bp + 16);
            }
            for (int mt = 0; mt < 4; ++mt)
                for (int nt = 0; nt < 2; ++nt)
                    acc[mt][nt] = __builtin_amdgcn_wmma_f32_16x16x32_bf16(
                        false, af[mt], false, bfr[nt], (short)0, acc[mt][nt],
                        false, false);
        }
    }

    // C/D layout: VGPR r -> M = r (lanes 0-15) / 8+r (lanes 16-31), N = l16.
    for (int mt = 0; mt < 4; ++mt)
        for (int nt = 0; nt < 2; ++nt)
            for (int r = 0; r < 8; ++r) {
                int row = m0 + rw * 64 + mt * 16 + half * 8 + r;
                int col = n0 + cw * 32 + nt * 16 + l16;
                size_t idx = (size_t)row * Nn + col;
                float v = acc[mt][nt][r];
                if (EPI == 0) {
                    Cf[idx] = v;
                } else if (EPI == 1) {
                    if (bias) v += bias[col];
                    Cf[idx] = v + resid[idx];
                } else {
                    v += bias[col];
                    Cb[idx] = (bf16)(v * sigmoidf_(v));
                }
            }
}

// ---------------------------------------------------------------------------
// kv state:  kv[b,h,d,e] = sum_n k[b,n,h,d] * v[b,n,h,e];  ksum[b,h,d] = sum_n k.
// Split-N across 32 blocks per (b,h); fp32 atomic accumulation. Tiny FLOPs.
// ---------------------------------------------------------------------------
__global__ __launch_bounds__(1024)
void kv_accum_kernel(const bf16* __restrict__ kr, const bf16* __restrict__ vb,
                     float* __restrict__ kv, float* __restrict__ ksum) {
    int t = threadIdx.x;
    int d = t >> 5, e = t & 31;
    int ns = blockIdx.x, h = blockIdx.y, b = blockIdx.z;
    int nbeg = ns * (NTOK / 32);
    float acc = 0.f, ks = 0.f;
    for (int n = nbeg; n < nbeg + NTOK / 32; ++n) {
        size_t ro = ((size_t)b * NTOK + n) * CCH + h * HDIM;
        float kf = (float)kr[ro + d];
        float vf = (float)vb[ro + e];
        acc += kf * vf;
        if (e == 0) ks += kf;
    }
    atomicAdd(&kv[(((size_t)b * HEADS + h) * HDIM + d) * HDIM + e], acc);
    if (e == 0) atomicAdd(&ksum[((size_t)b * HEADS + h) * HDIM + d], ks);
}

// ---------------------------------------------------------------------------
// num = q @ kv (WMMA, K=32 in one step), den = q . ksum + 1e-8,
// attn = num / den  -> bf16 in (B,N,C) layout. Block = 8 waves, 128 rows.
// ---------------------------------------------------------------------------
#define LDK 40
__global__ __launch_bounds__(256)
void attn_num_kernel(const bf16* __restrict__ qr, const float* __restrict__ kv,
                     const float* __restrict__ ksum, bf16* __restrict__ attn) {
    __shared__ bf16 kvT[32 * LDK];   // kvT[e][d] so B-fragments are contiguous in d
    __shared__ float ksum_s[32];
    __shared__ float denS[128];
    int t = threadIdx.x, wave = t >> 5, lane = t & 31;
    int half = lane >> 4, l16 = lane & 15;
    int m0 = blockIdx.x * 128, h = blockIdx.y, b = blockIdx.z;

    const float* kvb = kv + ((size_t)b * HEADS + h) * HDIM * HDIM;
    for (int i = 0; i < 4; ++i) {
        int idx = t + i * 256;
        int d = idx >> 5, e = idx & 31;
        kvT[e * LDK + d] = (bf16)kvb[idx];
    }
    if (t < 32) ksum_s[t] = ksum[((size_t)b * HEADS + h) * HDIM + t];
    __syncthreads();

    int rowL = m0 + wave * 16 + l16;
    const bf16* qbase = qr + ((size_t)b * NTOK + rowL) * CCH + h * HDIM;

    v16bf a, b0, b1;
    ((bfx8*)&a)[0] = *(const bfx8*)(qbase + half * 8);
    ((bfx8*)&a)[1] = *(const bfx8*)(qbase + 16 + half * 8);
    {
        const bf16* bp = &kvT[l16 * LDK + half * 8];
        ((bfx8*)&b0)[0] = *(const bfx8*)bp;
        ((bfx8*)&b0)[1] = *(const bfx8*)(bp + 16);
        bp = &kvT[(16 + l16) * LDK + half * 8];
        ((bfx8*)&b1)[0] = *(const bfx8*)bp;
        ((bfx8*)&b1)[1] = *(const bfx8*)(bp + 16);
    }
    v8f acc0, acc1;
    for (int e = 0; e < 8; ++e) { acc0[e] = 0.f; acc1[e] = 0.f; }
    acc0 = __builtin_amdgcn_wmma_f32_16x16x32_bf16(false, a, false, b0, (short)0, acc0, false, false);
    acc1 = __builtin_amdgcn_wmma_f32_16x16x32_bf16(false, a, false, b1, (short)0, acc1, false, false);

    float den = 1e-8f;
    for (int dd = 0; dd < HDIM; ++dd) den += (float)qbase[dd] * ksum_s[dd];
    if (half == 0) denS[wave * 16 + l16] = den;
    __syncthreads();

    for (int r = 0; r < 8; ++r) {
        int row = m0 + wave * 16 + half * 8 + r;
        float rd = 1.f / denS[wave * 16 + half * 8 + r];
        size_t ob = ((size_t)b * NTOK + row) * CCH + h * HDIM;
        attn[ob + l16]      = (bf16)(acc0[r] * rd);
        attn[ob + 16 + l16] = (bf16)(acc1[r] * rd);
    }
}

// ---------------------------------------------------------------------------
// Fused 3x3 depthwise conv (+bias) and GLU: out = (conv_a+b_a) * silu(conv_g+b_g)
// Input already has SiLU applied (inv-GEMM epilogue). bf16 in/out, 4 ch / thread.
// ---------------------------------------------------------------------------
__global__ __launch_bounds__(256)
void dwglu_kernel(const bf16* __restrict__ cin, const float* __restrict__ dww,
                  const float* __restrict__ dwb, bf16* __restrict__ outg) {
    unsigned gid = blockIdx.x * 256u + threadIdx.x;
    int c4 = gid & 1023;
    int x  = (gid >> 10) & 63;
    int y  = (gid >> 16) & 63;
    int b  = gid >> 22;
    int c  = c4 * 4;
    float a[4], g[4];
    for (int j = 0; j < 4; ++j) { a[j] = dwb[c + j]; g[j] = dwb[HID + c + j]; }
    for (int dy = -1; dy <= 1; ++dy) {
        int yy = y + dy; if (yy < 0 || yy > 63) continue;
        for (int dx = -1; dx <= 1; ++dx) {
            int xx = x + dx; if (xx < 0 || xx > 63) continue;
            const bf16* p = cin + (((size_t)(b * HHH + yy)) * WWW + xx) * (2 * HID);
            bfx4 av = *(const bfx4*)(p + c);
            bfx4 gv = *(const bfx4*)(p + HID + c);
            const float* wr = dww + (size_t)((dy + 1) * 3 + (dx + 1)) * (2 * HID);
            for (int j = 0; j < 4; ++j) {
                a[j] += (float)av[j] * wr[c + j];
                g[j] += (float)gv[j] * wr[HID + c + j];
            }
        }
    }
    bfx4 o;
    for (int j = 0; j < 4; ++j) o[j] = (bf16)(a[j] * g[j] * sigmoidf_(g[j]));
    *(bfx4*)&outg[(((size_t)(b * HHH + y)) * WWW + x) * HID + c] = o;
}

// ---------------------------------------------------------------------------
// Host-side orchestration
// ---------------------------------------------------------------------------
extern "C" void kernel_launch(void* const* d_in, const int* in_sizes, int n_in,
                              void* d_out, int out_size, void* d_ws, size_t ws_size,
                              hipStream_t stream) {
    (void)in_sizes; (void)n_in; (void)out_size; (void)ws_size;
    const float* x     = (const float*)d_in[0];
    const float* n1w   = (const float*)d_in[1];
    const float* qkvW  = (const float*)d_in[2];
    const float* qnw   = (const float*)d_in[3];
    const float* knw   = (const float*)d_in[4];
    const float* projW = (const float*)d_in[5];
    const float* projB = (const float*)d_in[6];
    const float* n2w   = (const float*)d_in[7];
    const float* invW  = (const float*)d_in[8];
    const float* invB  = (const float*)d_in[9];
    const float* dwW   = (const float*)d_in[10];
    const float* dwB   = (const float*)d_in[11];
    const float* pwW   = (const float*)d_in[12];
    float* out = (float*)d_out;

    // ---- workspace layout (~353 MB, regions reused across lifetimes) ----
    char* ws = (char*)d_ws;
    size_t off = 0;
    auto alloc = [&](size_t bytes) -> char* {
        char* p = ws + off;
        off += (bytes + 255) & ~(size_t)255;
        return p;
    };
    bf16*  qkvW_bf  = (bf16*)alloc((size_t)3 * CCH * CCH * 2);
    bf16*  projW_bf = (bf16*)alloc((size_t)CCH * CCH * 2);
    bf16*  invW_bf  = (bf16*)alloc((size_t)2 * HID * CCH * 2);
    bf16*  pwW_bf   = (bf16*)alloc((size_t)CCH * HID * 2);
    float* x1       = (float*)alloc((size_t)MROWS * CCH * 4);
    float* kvf      = (float*)alloc((size_t)BBATCH * HEADS * HDIM * HDIM * 4);
    float* ksumf    = (float*)alloc((size_t)BBATCH * HEADS * HDIM * 4);
    char*  regionA  = alloc((size_t)MROWS * 2 * HID * 2);   // max(h1, conv_in)
    char*  regionB  = alloc((size_t)MROWS * 3 * CCH * 4);   // qkv fp32, reused
    bf16*  qr       = (bf16*)alloc((size_t)MROWS * CCH * 2);
    bf16*  kr       = (bf16*)alloc((size_t)MROWS * CCH * 2);
    bf16*  vb       = (bf16*)alloc((size_t)MROWS * CCH * 2);

    bf16*  h1   = (bf16*)regionA;                                 // dead after qkv GEMM
    bf16*  cin  = (bf16*)regionA;                                 // conv input (reuse)
    float* qkvO = (float*)regionB;                                // dead after qknorm
    bf16*  attn = (bf16*)regionB;                                 // B + 0        (16.8 MB)
    bf16*  h2   = (bf16*)(regionB + (size_t)MROWS * CCH * 2);     // B + 16.8 MB
    bf16*  glu  = (bf16*)(regionB + (size_t)2 * MROWS * CCH * 2); // B + 33.6 MB  (67 MB)

    // 1. weight conversions to bf16
    f32_to_bf16_kernel<<<2048, 256, 0, stream>>>(qkvW, qkvW_bf, (long)3 * CCH * CCH);
    f32_to_bf16_kernel<<<2048, 256, 0, stream>>>(projW, projW_bf, (long)CCH * CCH);
    f32_to_bf16_kernel<<<4096, 256, 0, stream>>>(invW, invW_bf, (long)2 * HID * CCH);
    f32_to_bf16_kernel<<<4096, 256, 0, stream>>>(pwW, pwW_bf, (long)CCH * HID);

    // 2. attention branch
    rmsnorm_bf16_kernel<<<MROWS, 256, 0, stream>>>(x, n1w, h1);
    gemm_bf16_wmma<0><<<dim3(3 * CCH / TBN, MROWS / TBM), 256, 0, stream>>>(
        h1, qkvW_bf, nullptr, nullptr, qkvO, nullptr, MROWS, 3 * CCH, CCH);
    qknorm_kernel<<<MROWS, 256, 0, stream>>>(qkvO, qnw, knw, qr, kr, vb);

    hipMemsetAsync(kvf, 0, (size_t)BBATCH * HEADS * HDIM * HDIM * 4, stream);
    hipMemsetAsync(ksumf, 0, (size_t)BBATCH * HEADS * HDIM * 4, stream);
    kv_accum_kernel<<<dim3(32, HEADS, BBATCH), 1024, 0, stream>>>(kr, vb, kvf, ksumf);
    attn_num_kernel<<<dim3(NTOK / 128, HEADS, BBATCH), 256, 0, stream>>>(qr, kvf, ksumf, attn);

    // x1 = x + attn @ proj_w^T + proj_b
    gemm_bf16_wmma<1><<<dim3(CCH / TBN, MROWS / TBM), 256, 0, stream>>>(
        attn, projW_bf, projB, x, x1, nullptr, MROWS, CCH, CCH);

    // 3. GLUMBConv branch
    rmsnorm_bf16_kernel<<<MROWS, 256, 0, stream>>>(x1, n2w, h2);
    gemm_bf16_wmma<2><<<dim3(2 * HID / TBN, MROWS / TBM), 256, 0, stream>>>(
        h2, invW_bf, invB, nullptr, nullptr, cin, MROWS, 2 * HID, CCH);
    dwglu_kernel<<<(BBATCH * HHH * WWW * (HID / 4)) / 256, 256, 0, stream>>>(
        cin, dwW, dwB, glu);

    // out = x1 + glu @ pw_w^T
    gemm_bf16_wmma<1><<<dim3(CCH / TBN, MROWS / TBM), 256, 0, stream>>>(
        glu, pwW_bf, nullptr, x1, out, nullptr, MROWS, CCH, HID);
}
